// StableAudioAttnProcessor2_0_rotary_618475290954
// MI455X (gfx1250) — matrix operand order repaired and checked by reading, not compiled, not run on hardware
//
#include <hip/hip_runtime.h>
#include <hip/hip_bf16.h>

typedef __bf16 bf16;
typedef __attribute__((ext_vector_type(16))) __bf16 v16bf;
typedef __attribute__((ext_vector_type(8)))  __bf16 v8bf;
typedef __attribute__((ext_vector_type(4)))  __bf16 v4bf;
typedef __attribute__((ext_vector_type(8)))  float  v8f;

union AB16 { v16bf v; v8bf h[2]; bf16 e[16]; };

#define DEV __device__ __forceinline__

static constexpr int Bv   = 2;
static constexpr int Sq   = 2048;
static constexpr int C1   = 1536;
static constexpr int CE   = 768;
static constexpr int Hh   = 24;
static constexpr int KVHh = 12;
static constexpr int Dd   = 64;
static constexpr int Tenc = 512;
static constexpr int Tip  = 1024;

static constexpr float LN1E4_OVER_32 = 9.210340371976184f / 32.0f;

DEV v8f wmma_bf(const AB16& a, const AB16& b, v8f c) {
    return __builtin_amdgcn_wmma_f32_16x16x32_bf16(
        false, a.v, false, b.v, (short)0, c, false, false);
}

DEV float hmax16(float x) {
    #pragma unroll
    for (int off = 1; off < 16; off <<= 1) x = fmaxf(x, __shfl_xor(x, off, 32));
    return x;
}

// ---------------- elementwise converts ----------------

__global__ void cvt_bf16(const float* __restrict__ in, bf16* __restrict__ out, int n) {
    int i = blockIdx.x * blockDim.x + threadIdx.x;
    if (i < n) out[i] = (bf16)in[i];
}

// out[N][K] = in[K][N]  (weight transpose to N-major for WMMA B operand)
__global__ void cvt_t_bf16(const float* __restrict__ in, bf16* __restrict__ out, int K, int N) {
    int i = blockIdx.x * blockDim.x + threadIdx.x;
    if (i >= N * K) return;
    int nn = i / K, kk = i - nn * K;
    out[i] = (bf16)in[(size_t)kk * N + nn];
}

// ---------------- reshape + RoPE ----------------

// qproj [B*S][1536] -> qn [B,H,S,D] (no rope) and qr [B,H,S,D] (rope pos = s*0.5)
// Both pre-scaled by 1/sqrt(D) so attention uses raw WMMA results.
__global__ void reshape_rope_q(const bf16* __restrict__ qp, bf16* __restrict__ qn,
                               bf16* __restrict__ qr, int n) {
    int i = blockIdx.x * blockDim.x + threadIdx.x;
    if (i >= n) return;
    int d = i & 63;
    int s = (i >> 6) % Sq;
    int h = ((i >> 6) / Sq) % Hh;
    int b = i / (64 * Sq * Hh);
    size_t base = ((size_t)(b * Sq + s)) * C1 + h * 64;
    float x = (float)qp[base + d];
    qn[i] = (bf16)(0.125f * x);
    int idx = d & 31;
    float fr = ((float)s * 0.5f) * __expf(-(float)idx * LN1E4_OVER_32);
    int dp = (d < 32) ? (2 * d + 1) : (2 * (d - 32));
    float xp = (float)qp[base + dp];
    float xr = (d < 32) ? -xp : xp;
    qr[i] = (bf16)(0.125f * (x * __cosf(fr) + xr * __sinf(fr)));
}

// kproj/vproj [B*512][768] -> K [B,KVH,Tenc,D], V^T [B,KVH,D,Tenc]
__global__ void reshape_kv(const bf16* __restrict__ kp, const bf16* __restrict__ vp,
                           bf16* __restrict__ kk, bf16* __restrict__ vt, int n) {
    int i = blockIdx.x * blockDim.x + threadIdx.x;
    if (i >= n) return;
    int d = i & 63;
    int s = (i >> 6) % Tenc;
    int h = ((i >> 6) / Tenc) % KVHh;
    int b = i / (64 * Tenc * KVHh);
    size_t src = ((size_t)(b * Tenc + s)) * CE + h * 64 + d;
    kk[i] = kp[src];
    vt[(((size_t)(b * KVHh + h)) * Dd + d) * Tenc + s] = vp[src];
}

// ipkproj/ipvproj [B*512][1536] -> roped K [B,KVH,Tip,D], roped V^T [B,KVH,D,Tip]
__global__ void reshape_rope_ip(const bf16* __restrict__ kp, const bf16* __restrict__ vp,
                                bf16* __restrict__ ipk, bf16* __restrict__ ipvt, int n) {
    int i = blockIdx.x * blockDim.x + threadIdx.x;
    if (i >= n) return;
    int d  = i & 63;
    int tk = (i >> 6) % Tip;
    int h  = ((i >> 6) / Tip) % KVHh;
    int b  = i / (64 * Tip * KVHh);
    int s = tk >> 1;
    size_t rowb = ((size_t)(b * 512 + s)) * C1 + (tk & 1) * 768 + h * 64;
    int idx = d & 31;
    float fr = (float)tk * __expf(-(float)idx * LN1E4_OVER_32);
    float c = __cosf(fr), sn = __sinf(fr);
    int dp = (d < 32) ? (2 * d + 1) : (2 * (d - 32));
    float sgn = (d < 32) ? -1.0f : 1.0f;
    float kx = (float)kp[rowb + d], kxp = (float)kp[rowb + dp];
    float vx = (float)vp[rowb + d], vxp = (float)vp[rowb + dp];
    ipk[i] = (bf16)(kx * c + sgn * kxp * sn);
    ipvt[(((size_t)(b * KVHh + h)) * Dd + d) * Tip + tk] = (bf16)(vx * c + sgn * vxp * sn);
}

// ---------------- generic bf16 WMMA GEMM: C[m,n] = sum_k A[m,k]*Bt[n,k] ----------------
// Wave computes 64x64 tile (16 accumulators): 16 b128 loads per 16 WMMAs.
// MODE 0: store bf16 | MODE 1: bf16 (acc + bf16 addend) | MODE 2: f32 (acc + f32 addend)

template <int MODE>
__global__ void __launch_bounds__(128) gemm_bt(
    const bf16* __restrict__ A, const bf16* __restrict__ Bt,
    void* __restrict__ Cout, const void* __restrict__ Add,
    int M, int N, int K) {
    const int lane = threadIdx.x & 31;
    const int hi   = lane >> 4;
    const int ln15 = lane & 15;
    const int wid  = blockIdx.x * (blockDim.x >> 5) + (threadIdx.x >> 5);
    const int tilesN = N >> 6;
    const int tilesM = M >> 6;
    if (wid >= tilesM * tilesN) return;
    const int tm = wid / tilesN;
    const int tn = wid - tm * tilesN;
    const int m0 = tm << 6;
    const int n0 = tn << 6;

    v8f acc[4][4];
    #pragma unroll
    for (int i = 0; i < 4; i++)
        #pragma unroll
        for (int j = 0; j < 4; j++) acc[i][j] = v8f{};

    const bf16* ap[4];
    const bf16* bp[4];
    #pragma unroll
    for (int i = 0; i < 4; i++)
        ap[i] = A + (size_t)(m0 + i * 16 + ln15) * K + hi * 8;
    #pragma unroll
    for (int j = 0; j < 4; j++)
        bp[j] = Bt + (size_t)(n0 + j * 16 + ln15) * K + hi * 8;

    for (int k0 = 0; k0 < K; k0 += 32) {
        AB16 a[4], b[4];
        #pragma unroll
        for (int i = 0; i < 4; i++) {
            a[i].h[0] = *(const v8bf*)(ap[i] + k0);
            a[i].h[1] = *(const v8bf*)(ap[i] + k0 + 16);
        }
        #pragma unroll
        for (int j = 0; j < 4; j++) {
            b[j].h[0] = *(const v8bf*)(bp[j] + k0);
            b[j].h[1] = *(const v8bf*)(bp[j] + k0 + 16);
        }
        #pragma unroll
        for (int i = 0; i < 4; i++)
            #pragma unroll
            for (int j = 0; j < 4; j++)
                acc[i][j] = wmma_bf(a[i], b[j], acc[i][j]);
    }

    #pragma unroll
    for (int i = 0; i < 4; i++) {
        #pragma unroll
        for (int j = 0; j < 4; j++) {
            const int col = n0 + j * 16 + ln15;
            #pragma unroll
            for (int r = 0; r < 8; r++) {
                const size_t idx = (size_t)(m0 + i * 16 + r + hi * 8) * N + col;
                float vsum = acc[i][j][r];
                if constexpr (MODE == 0) {
                    ((bf16*)Cout)[idx] = (bf16)vsum;
                } else if constexpr (MODE == 1) {
                    ((bf16*)Cout)[idx] = (bf16)(vsum + (float)((const bf16*)Add)[idx]);
                } else {
                    ((float*)Cout)[idx] = vsum + ((const float*)Add)[idx];
                }
            }
        }
    }
}

// ---------------- flash attention (bf16 WMMA, fp32 online softmax) ----------------
// Q [B,H,S,D] (pre-scaled by 1/8), K [B,KVH,T,D], Vt [B,KVH,D,T], Out [B*S][H*D] bf16.
// One wave per 16-row q tile; 64 keys per block; key interleave: S-tile j holds keys 4c+j
// so each lane's 4 probabilities per row are adjacent keys -> one b64 LDS store.
// Row-sum computed by WMMA against an all-ones B matrix (no shuffle reduction).

__global__ void __launch_bounds__(128) attn_fa(
    const bf16* __restrict__ Q, const bf16* __restrict__ Kp,
    const bf16* __restrict__ Vt, bf16* __restrict__ Out, int T) {
    __shared__ __align__(16) bf16 plds[4][16 * 64];
    const int lane = threadIdx.x & 31;
    const int hi   = lane >> 4;
    const int ln15 = lane & 15;
    const int w    = threadIdx.x >> 5;
    const int gw   = blockIdx.x * 4 + w;
    const int QT   = Sq / 16;
    const int qt = gw % QT;
    const int h  = (gw / QT) % Hh;
    const int b  = gw / (QT * Hh);

    const bf16* qbase = Q  + (((size_t)(b * Hh + h)) * Sq + qt * 16) * Dd;
    const bf16* kbase = Kp + ((size_t)(b * KVHh + (h >> 1))) * (size_t)T * Dd;
    const bf16* vbase = Vt + ((size_t)(b * KVHh + (h >> 1))) * (size_t)Dd * T;
    bf16* plw = plds[w];

    // Q tile 16x64 in two A-layout chunks (k = d 0..31, 32..63)
    AB16 qa0, qa1;
    const bf16* qrow = qbase + (size_t)ln15 * Dd + hi * 8;
    qa0.h[0] = *(const v8bf*)(qrow + 0);
    qa0.h[1] = *(const v8bf*)(qrow + 16);
    qa1.h[0] = *(const v8bf*)(qrow + 32);
    qa1.h[1] = *(const v8bf*)(qrow + 48);

    AB16 onesb;
    #pragma unroll
    for (int i = 0; i < 16; i++) onesb.e[i] = (bf16)1.0f;

    v8f o[4] = {v8f{}, v8f{}, v8f{}, v8f{}};
    v8f lacc = v8f{};
    float mrow[8];
    #pragma unroll
    for (int r = 0; r < 8; r++) mrow[r] = -3.0e38f;

    for (int t0 = 0; t0 < T; t0 += 64) {
        // ---- S = Q*K^T: 4 tiles, tile j holds keys t0 + 4*c + j ----
        v8f s[4];
        #pragma unroll
        for (int j = 0; j < 4; j++) {
            const bf16* kr = kbase + (size_t)(t0 + 4 * ln15 + j) * Dd + hi * 8;
            AB16 bk;
            bk.h[0] = *(const v8bf*)(kr + 0);
            bk.h[1] = *(const v8bf*)(kr + 16);
            s[j] = wmma_bf(qa0, bk, v8f{});
            bk.h[0] = *(const v8bf*)(kr + 32);
            bk.h[1] = *(const v8bf*)(kr + 48);
            s[j] = wmma_bf(qa1, bk, s[j]);
        }
        // ---- online softmax (rows live in half-wave groups) ----
        #pragma unroll
        for (int r = 0; r < 8; r++) {
            float mx = fmaxf(fmaxf(s[0][r], s[1][r]), fmaxf(s[2][r], s[3][r]));
            mx = hmax16(mx);
            float mnew = fmaxf(mrow[r], mx);
            float al = __expf(mrow[r] - mnew);
            mrow[r] = mnew;
            float e0 = __expf(s[0][r] - mnew);
            float e1 = __expf(s[1][r] - mnew);
            float e2 = __expf(s[2][r] - mnew);
            float e3 = __expf(s[3][r] - mnew);
            lacc[r] *= al;
            o[0][r] *= al; o[1][r] *= al; o[2][r] *= al; o[3][r] *= al;
            v4bf pk;
            pk[0] = (bf16)e0; pk[1] = (bf16)e1; pk[2] = (bf16)e2; pk[3] = (bf16)e3;
            *(v4bf*)(plw + (r + hi * 8) * 64 + 4 * ln15) = pk;
        }
        // ---- re-load P in WMMA A layout (same-wave LDS, in order) ----
        AB16 pa0, pa1;
        const bf16* prow = plw + ln15 * 64 + hi * 8;
        pa0.h[0] = *(const v8bf*)(prow + 0);
        pa0.h[1] = *(const v8bf*)(prow + 16);
        pa1.h[0] = *(const v8bf*)(prow + 32);
        pa1.h[1] = *(const v8bf*)(prow + 48);
        // ---- row sums via WMMA against ones ----
        lacc = wmma_bf(pa0, onesb, lacc);
        lacc = wmma_bf(pa1, onesb, lacc);
        // ---- O += P*V : V^T rows are contiguous key runs ----
        #pragma unroll
        for (int t = 0; t < 4; t++) {
            const bf16* vr = vbase + (size_t)(t * 16 + ln15) * T + t0 + hi * 8;
            AB16 bv;
            bv.h[0] = *(const v8bf*)(vr + 0);
            bv.h[1] = *(const v8bf*)(vr + 16);
            o[t] = wmma_bf(pa0, bv, o[t]);
            bv.h[0] = *(const v8bf*)(vr + 32);
            bv.h[1] = *(const v8bf*)(vr + 48);
            o[t] = wmma_bf(pa1, bv, o[t]);
        }
    }

    // ---- normalize and store ----
    #pragma unroll
    for (int r = 0; r < 8; r++) {
        float inv = 1.0f / lacc[r];
        size_t rowo = (size_t)(b * Sq + qt * 16 + r + hi * 8) * (Hh * Dd) + h * Dd + ln15;
        #pragma unroll
        for (int t = 0; t < 4; t++)
            Out[rowo + t * 16] = (bf16)(o[t][r] * inv);
    }
}

// ---------------- host orchestration ----------------

extern "C" void kernel_launch(void* const* d_in, const int* in_sizes, int n_in,
                              void* d_out, int out_size, void* d_ws, size_t ws_size,
                              hipStream_t stream) {
    const float* hs   = (const float*)d_in[0];
    const float* enc  = (const float*)d_in[1];
    const float* encc = (const float*)d_in[2];
    const float* Wq   = (const float*)d_in[3];
    const float* Wk   = (const float*)d_in[4];
    const float* Wv   = (const float*)d_in[5];
    const float* Wkip = (const float*)d_in[6];
    const float* Wvip = (const float*)d_in[7];
    const float* Wcv  = (const float*)d_in[8];
    const float* Wout = (const float*)d_in[9];
    float* outp = (float*)d_out;

    const int Mq = Bv * Sq;               // 4096
    const int Me = Bv * 512;              // 1024
    char* ws = (char*)d_ws;
    size_t off = 0;
    auto alloc_bf = [&](size_t elems) -> bf16* {
        bf16* p = (bf16*)(ws + off);
        off += ((elems * sizeof(bf16) + 255) / 256) * 256;
        return p;
    };

    bf16* hsb    = alloc_bf((size_t)Mq * C1);
    bf16* encb   = alloc_bf((size_t)Me * CE);
    bf16* enccb  = alloc_bf((size_t)Me * CE);
    bf16* WqT    = alloc_bf((size_t)C1 * C1);
    bf16* WkT    = alloc_bf((size_t)CE * CE);
    bf16* WvT    = alloc_bf((size_t)CE * CE);
    bf16* WkipT  = alloc_bf((size_t)C1 * CE);
    bf16* WvipT  = alloc_bf((size_t)C1 * CE);
    bf16* Wcvb   = alloc_bf((size_t)C1 * C1);
    bf16* WoutT  = alloc_bf((size_t)C1 * C1);
    bf16* qproj  = alloc_bf((size_t)Mq * C1);
    bf16* kproj  = alloc_bf((size_t)Me * CE);
    bf16* vproj  = alloc_bf((size_t)Me * CE);
    bf16* ikproj = alloc_bf((size_t)Me * C1);
    bf16* ivproj = alloc_bf((size_t)Me * C1);
    bf16* qn     = alloc_bf((size_t)Bv * Hh * Sq * Dd);
    bf16* qr     = alloc_bf((size_t)Bv * Hh * Sq * Dd);
    bf16* kk     = alloc_bf((size_t)Bv * KVHh * Tenc * Dd);
    bf16* vt     = alloc_bf((size_t)Bv * KVHh * Dd * Tenc);
    bf16* ipk    = alloc_bf((size_t)Bv * KVHh * Tip * Dd);
    bf16* ipvt   = alloc_bf((size_t)Bv * KVHh * Dd * Tip);
    bf16* hsA    = alloc_bf((size_t)Mq * C1);
    bf16* iphsA  = alloc_bf((size_t)Mq * C1);
    bf16* ybuf   = alloc_bf((size_t)Mq * C1);
    (void)ws_size; (void)in_sizes; (void)n_in; (void)out_size;

    const int TH = 256;
    auto gr = [](int n, int t) { return (n + t - 1) / t; };

    // converts
    cvt_bf16<<<gr(Mq * C1, TH), TH, 0, stream>>>(hs, hsb, Mq * C1);
    cvt_bf16<<<gr(Me * CE, TH), TH, 0, stream>>>(enc, encb, Me * CE);
    cvt_bf16<<<gr(Me * CE, TH), TH, 0, stream>>>(encc, enccb, Me * CE);
    cvt_t_bf16<<<gr(C1 * C1, TH), TH, 0, stream>>>(Wq, WqT, C1, C1);
    cvt_t_bf16<<<gr(CE * CE, TH), TH, 0, stream>>>(Wk, WkT, CE, CE);
    cvt_t_bf16<<<gr(CE * CE, TH), TH, 0, stream>>>(Wv, WvT, CE, CE);
    cvt_t_bf16<<<gr(CE * C1, TH), TH, 0, stream>>>(Wkip, WkipT, CE, C1);
    cvt_t_bf16<<<gr(CE * C1, TH), TH, 0, stream>>>(Wvip, WvipT, CE, C1);
    cvt_bf16<<<gr(C1 * C1, TH), TH, 0, stream>>>(Wcv, Wcvb, C1 * C1);
    cvt_t_bf16<<<gr(C1 * C1, TH), TH, 0, stream>>>(Wout, WoutT, C1, C1);

    // projections (wave = 64x64 tile, 4 waves/block)
    auto gemm_blocks = [](int M, int N) { return ((M / 64) * (N / 64) + 3) / 4; };
    gemm_bt<0><<<gemm_blocks(Mq, C1), 128, 0, stream>>>(hsb,   WqT,   qproj,  nullptr, Mq, C1, C1);
    gemm_bt<0><<<gemm_blocks(Me, CE), 128, 0, stream>>>(encb,  WkT,   kproj,  nullptr, Me, CE, CE);
    gemm_bt<0><<<gemm_blocks(Me, CE), 128, 0, stream>>>(encb,  WvT,   vproj,  nullptr, Me, CE, CE);
    gemm_bt<0><<<gemm_blocks(Me, C1), 128, 0, stream>>>(enccb, WkipT, ikproj, nullptr, Me, C1, CE);
    gemm_bt<0><<<gemm_blocks(Me, C1), 128, 0, stream>>>(enccb, WvipT, ivproj, nullptr, Me, C1, CE);

    // reshape + RoPE
    int nq = Bv * Hh * Sq * Dd;
    reshape_rope_q<<<gr(nq, TH), TH, 0, stream>>>(qproj, qn, qr, nq);
    int nkv = Bv * KVHh * Tenc * Dd;
    reshape_kv<<<gr(nkv, TH), TH, 0, stream>>>(kproj, vproj, kk, vt, nkv);
    int nip = Bv * KVHh * Tip * Dd;
    reshape_rope_ip<<<gr(nip, TH), TH, 0, stream>>>(ikproj, ivproj, ipk, ipvt, nip);

    // flash attention: B*H*(S/16) waves, 4 waves/block
    int attnBlocks = (Bv * Hh * (Sq / 16)) / 4;  // 1536
    attn_fa<<<attnBlocks, 128, 0, stream>>>(qn, kk, vt, hsA, Tenc);
    attn_fa<<<attnBlocks, 128, 0, stream>>>(qr, ipk, ipvt, iphsA, Tip);

    // y = hs + ip_hs @ Wconv^T  (Wconv already N x K)
    gemm_bt<1><<<gemm_blocks(Mq, C1), 128, 0, stream>>>(iphsA, Wcvb, ybuf, hsA, Mq, C1, C1);
    // out = y @ Wout + residual (fp32)
    gemm_bt<2><<<gemm_blocks(Mq, C1), 128, 0, stream>>>(ybuf, WoutT, outp, hs, Mq, C1, C1);
}
